// CapsuleNetwork_24489903522270
// MI455X (gfx1250) — compile-verified, matrix operand order checked by reading, not run on hardware
//
#include <hip/hip_runtime.h>
#include <math.h>

// ---------- types ----------
typedef __bf16 bf16;
typedef __attribute__((ext_vector_type(16))) __bf16 v16bf;
typedef __attribute__((ext_vector_type(8)))  __bf16 v8bf;
typedef __attribute__((ext_vector_type(8)))  float  v8f;

__device__ __forceinline__ bf16 to_bf16(float f) {
  unsigned u = __builtin_bit_cast(unsigned, f);
  unsigned r = (u + 0x7FFFu + ((u >> 16) & 1u)) >> 16;
  unsigned short h = (unsigned short)r;
  return __builtin_bit_cast(bf16, h);
}
__device__ __forceinline__ float from_bf16(bf16 h) {
  unsigned short s = __builtin_bit_cast(unsigned short, h);
  unsigned u = ((unsigned)s) << 16;
  return __builtin_bit_cast(float, u);
}
__device__ __forceinline__ v8f zero_v8f() {
  v8f z;
#pragma unroll
  for (int i = 0; i < 8; ++i) z[i] = 0.f;
  return z;
}
__device__ __forceinline__ float sigmoidf(float x) { return 1.f / (1.f + __expf(-x)); }

// 16x32 bf16 A/B fragment load (row-major source, row stride ld elements).
// ISA layout: lanes 0-15 -> row = row0+lane, K {0..7} in v0..3, {16..23} in v4..7
//             lanes 16-31 -> same rows, K {8..15} and {24..31}.
__device__ __forceinline__ v16bf load_frag(const bf16* __restrict__ base, int row0, int k0, int ld) {
  int lane = threadIdx.x & 31;
  const bf16* p = base + (size_t)(row0 + (lane & 15)) * ld + k0 + ((lane >> 4) << 3);
  v8bf lo = *(const v8bf*)(p);
  v8bf hi = *(const v8bf*)(p + 16);
  v16bf f;
#pragma unroll
  for (int i = 0; i < 8; ++i) { f[i] = lo[i]; f[i + 8] = hi[i]; }
  return f;
}

// ---------- constants ----------
#define Bn 256
#define Tn 64
#define Dn 300
#define Dp 320
#define Hn 512
#define G4H 2048
#define KCAT 832
#define DAn 256
#define Rn 8
#define Cn 64
#define Pn 16
#define H2 1024

// ---------- prep kernels ----------
__global__ void embed_gather(const int* __restrict__ ix, const int* __restrict__ lens,
                             const float* __restrict__ emb, bf16* __restrict__ X, int backward) {
  int gid = blockIdx.x * 256 + threadIdx.x;       // (t, b, d) over T*B*320
  if (gid >= Tn * Bn * Dp) return;
  int d = gid % Dp;
  int bt = gid / Dp;
  int b = bt & (Bn - 1);
  int t = bt >> 8;
  int tt = t;
  if (backward) { int len = lens[b]; tt = (t < len) ? (len - 1 - t) : t; }
  int tok = ix[b * Tn + tt];
  float v = (d < Dn) ? emb[(size_t)tok * Dn + d] : 0.f;
  X[gid] = to_bf16(v);
}

__global__ void pack_wcat(const float* __restrict__ wih, const float* __restrict__ whh,
                          const float* __restrict__ bih, const float* __restrict__ bhh,
                          bf16* __restrict__ W, float* __restrict__ bias) {
  int gid = blockIdx.x * 256 + threadIdx.x;       // 2048*832
  if (gid >= G4H * KCAT) return;
  int n = gid / KCAT, k = gid % KCAT;
  float v = (k < Dp) ? ((k < Dn) ? wih[n * Dn + k] : 0.f) : whh[n * Hn + (k - Dp)];
  W[gid] = to_bf16(v);
  if (gid < G4H) bias[gid] = bih[gid] + bhh[gid];
}

__global__ void pack_ws1(const float* __restrict__ ws1, bf16* __restrict__ W1b) {
  int gid = blockIdx.x * 256 + threadIdx.x;       // 256*1024
  if (gid < DAn * H2) W1b[gid] = to_bf16(ws1[gid]);
}

__global__ void caps_transpose(const float* __restrict__ caps, bf16* __restrict__ capsT) {
  int gid = blockIdx.x * 256 + threadIdx.x;       // (r,k,h) over 8*1024*1024
  if (gid >= Rn * H2 * H2) return;
  int h = gid & 1023, k = (gid >> 10) & 1023, r = gid >> 20;
  capsT[gid] = to_bf16(caps[(((size_t)r * H2 + h) << 10) + k]);
}

__global__ void zero_state(bf16* hf, bf16* hb, float* cf, float* cb) {
  int gid = blockIdx.x * 256 + threadIdx.x;       // 256*512
  if (gid >= Bn * Hn) return;
  bf16 z = to_bf16(0.f);
  hf[gid] = z; hb[gid] = z; cf[gid] = 0.f; cb[gid] = 0.f;
}

// ---------- LSTM step GEMM: Z(256x2048) = [X_t | H](256x832) x Wcat^T + bias ----------
// 4 waves/block; each wave owns a 32x64 tile (2 A frags x 4 B frags -> 8 WMMA / k-tile).
// K loop split at the X|H boundary (tile 10) so each loop is branch-free and pipelines.
__global__ __launch_bounds__(128)
void lstm_step_gemm(const bf16* __restrict__ Xt, const bf16* __restrict__ H,
                    const bf16* __restrict__ W, const float* __restrict__ bias,
                    float* __restrict__ Z) {
  int wave = threadIdx.x >> 5, lane = threadIdx.x & 31;
  int m0 = blockIdx.y * 32;
  int n0 = blockIdx.x * 256 + wave * 64;
  v8f acc[2][4];
#pragma unroll
  for (int i = 0; i < 2; ++i)
#pragma unroll
    for (int j = 0; j < 4; ++j) acc[i][j] = zero_v8f();

  // X part: k-tiles 0..9
  for (int kt = 0; kt < Dp / 32; ++kt) {
    int k0 = kt * 32;
    __builtin_prefetch(W + (size_t)(n0 + (lane & 15) * 4) * KCAT + k0 + 32, 0, 1);
    v16bf bfr[4];
#pragma unroll
    for (int j = 0; j < 4; ++j) bfr[j] = load_frag(W, n0 + j * 16, k0, KCAT);
    v16bf a0 = load_frag(Xt, m0, k0, Dp);
    v16bf a1 = load_frag(Xt, m0 + 16, k0, Dp);
#pragma unroll
    for (int j = 0; j < 4; ++j) {
      acc[0][j] = __builtin_amdgcn_wmma_f32_16x16x32_bf16(false, a0, false, bfr[j], (short)0,
                                                          acc[0][j], false, false);
      acc[1][j] = __builtin_amdgcn_wmma_f32_16x16x32_bf16(false, a1, false, bfr[j], (short)0,
                                                          acc[1][j], false, false);
    }
  }
  // H part: k-tiles 10..25
  for (int kt = 0; kt < Hn / 32; ++kt) {
    int k0 = kt * 32;
    if (kt + 1 < Hn / 32)
      __builtin_prefetch(W + (size_t)(n0 + (lane & 15) * 4) * KCAT + Dp + k0 + 32, 0, 1);
    v16bf bfr[4];
#pragma unroll
    for (int j = 0; j < 4; ++j) bfr[j] = load_frag(W, n0 + j * 16, Dp + k0, KCAT);
    v16bf a0 = load_frag(H, m0, k0, Hn);
    v16bf a1 = load_frag(H, m0 + 16, k0, Hn);
#pragma unroll
    for (int j = 0; j < 4; ++j) {
      acc[0][j] = __builtin_amdgcn_wmma_f32_16x16x32_bf16(false, a0, false, bfr[j], (short)0,
                                                          acc[0][j], false, false);
      acc[1][j] = __builtin_amdgcn_wmma_f32_16x16x32_bf16(false, a1, false, bfr[j], (short)0,
                                                          acc[1][j], false, false);
    }
  }
  int sel = lane >> 4, col = lane & 15;
#pragma unroll
  for (int j = 0; j < 4; ++j) {
    int n = n0 + j * 16 + col;
    float bs = bias[n];
#pragma unroll
    for (int i = 0; i < 2; ++i) {
#pragma unroll
      for (int e = 0; e < 8; ++e) {
        int m = m0 + i * 16 + e + sel * 8;
        Z[(size_t)m * G4H + n] = acc[i][j][e] + bs;
      }
    }
  }
}

// ---------- LSTM gate / state update ----------
__global__ __launch_bounds__(256)
void lstm_gate(const float* __restrict__ Z, const int* __restrict__ lens,
               float* __restrict__ cbuf, bf16* __restrict__ hbuf,
               bf16* __restrict__ outb, int t, int backward) {
  int gid = blockIdx.x * 256 + threadIdx.x;       // 256*512
  if (gid >= Bn * Hn) return;
  int b = gid >> 9, j = gid & (Hn - 1);
  int len = lens[b];
  bool m = (t < len);
  const float* zr = Z + (size_t)b * G4H;
  float si = sigmoidf(zr[j]);
  float sf = sigmoidf(zr[Hn + j]);
  float g  = tanhf(zr[2 * Hn + j]);
  float so = sigmoidf(zr[3 * Hn + j]);
  float c_old = cbuf[gid];
  float h_old = from_bf16(hbuf[gid]);
  float cn = sf * c_old + si * g;
  float hn = so * tanhf(cn);
  cbuf[gid] = m ? cn : c_old;
  hbuf[gid] = to_bf16(m ? hn : h_old);
  float y = m ? hn : 0.f;
  int p = t, off = 0;
  if (backward) { p = m ? (len - 1 - t) : t; off = Hn; }
  outb[((size_t)(b * Tn + p)) * H2 + off + j] = to_bf16(y);
}

// ---------- attention projection: U(16384x256) = tanh(out(16384x1024) x WS1^T) ----------
// each wave: 64x64 tile (4 A frags x 4 B frags -> 16 WMMA / k-tile)
__global__ __launch_bounds__(128)
void attn_proj_gemm(const bf16* __restrict__ A, const bf16* __restrict__ W,
                    bf16* __restrict__ U) {
  int wave = threadIdx.x >> 5, lane = threadIdx.x & 31;
  int m0 = blockIdx.x * 64;
  int n0 = wave * 64;
  v8f acc[4][4];
#pragma unroll
  for (int i = 0; i < 4; ++i)
#pragma unroll
    for (int j = 0; j < 4; ++j) acc[i][j] = zero_v8f();
  for (int kt = 0; kt < H2 / 32; ++kt) {
    int k0 = kt * 32;
    v16bf a[4];
#pragma unroll
    for (int i = 0; i < 4; ++i) a[i] = load_frag(A, m0 + i * 16, k0, H2);
#pragma unroll
    for (int j = 0; j < 4; ++j) {
      v16bf bm = load_frag(W, n0 + j * 16, k0, H2);
#pragma unroll
      for (int i = 0; i < 4; ++i)
        acc[i][j] = __builtin_amdgcn_wmma_f32_16x16x32_bf16(false, a[i], false, bm, (short)0,
                                                            acc[i][j], false, false);
    }
  }
  int sel = lane >> 4, col = lane & 15;
#pragma unroll
  for (int j = 0; j < 4; ++j) {
    int n = n0 + j * 16 + col;
#pragma unroll
    for (int i = 0; i < 4; ++i) {
#pragma unroll
      for (int e = 0; e < 8; ++e) {
        int m = m0 + i * 16 + e + sel * 8;
        U[(size_t)m * DAn + n] = to_bf16(tanhf(acc[i][j][e]));
      }
    }
  }
}

// ---------- scores: (16384 x 8) = U x ws2^T ----------
__global__ __launch_bounds__(256)
void score_kernel(const bf16* __restrict__ U, const float* __restrict__ ws2,
                  float* __restrict__ scores) {
  int gid = blockIdx.x * 256 + threadIdx.x;       // (row, r) over 16384*8
  if (gid >= Bn * Tn * Rn) return;
  int row = gid >> 3, r = gid & 7;
  const bf16* u = U + (size_t)row * DAn;
  const float* w = ws2 + r * DAn;
  float acc = 0.f;
#pragma unroll 8
  for (int k = 0; k < DAn; ++k) acc += from_bf16(u[k]) * w[k];
  scores[gid] = acc;
}

// ---------- softmax over T per (b,r): one wave per pair ----------
__global__ __launch_bounds__(256)
void att_softmax(const float* __restrict__ scores, float* __restrict__ att) {
  int wv = threadIdx.x >> 5, lane = threadIdx.x & 31;
  int pair = blockIdx.x * 8 + wv;                  // pair = b*8 + r
  int b = pair >> 3, r = pair & 7;
  float x0 = scores[(size_t)(b * Tn + lane) * Rn + r];
  float x1 = scores[(size_t)(b * Tn + lane + 32) * Rn + r];
  float m = fmaxf(x0, x1);
  for (int s = 16; s; s >>= 1) m = fmaxf(m, __shfl_xor(m, s, 32));
  float e0 = __expf(x0 - m), e1 = __expf(x1 - m);
  float sm = e0 + e1;
  for (int s = 16; s; s >>= 1) sm += __shfl_xor(sm, s, 32);
  att[(size_t)pair * Tn + lane] = e0 / sm;
  att[(size_t)pair * Tn + lane + 32] = e1 / sm;
}

// ---------- sem(r,b,h) = sum_t att[b,r,t] * out[b,t,h] ----------
__global__ __launch_bounds__(256)
void sem_kernel(const float* __restrict__ att, const bf16* __restrict__ outb,
                bf16* __restrict__ semb) {
  int pair = blockIdx.x;                           // b*8 + r
  int b = pair >> 3, r = pair & 7;
  int tid = threadIdx.x;
  float acc[4] = {0.f, 0.f, 0.f, 0.f};
  for (int t = 0; t < Tn; ++t) {
    float a = att[(size_t)pair * Tn + t];
    const bf16* op = outb + (size_t)(b * Tn + t) * H2;
#pragma unroll
    for (int j = 0; j < 4; ++j) acc[j] += a * from_bf16(op[tid + j * 256]);
  }
  bf16* dst = semb + ((size_t)r * Bn + b) * H2;
#pragma unroll
  for (int j = 0; j < 4; ++j) dst[tid + j * 256] = to_bf16(acc[j]);
}

// ---------- capsule projection: per r, pred = sem_r(256x1024) x capsT_r^T (1024x1024) ----------
// each wave: 64x64 tile
__global__ __launch_bounds__(128)
void caps_gemm(const bf16* __restrict__ semb, const bf16* __restrict__ capsT,
               float* __restrict__ pred) {
  int r = blockIdx.z;
  const bf16* A = semb + (size_t)r * Bn * H2;
  const bf16* Bm = capsT + (size_t)r * H2 * H2;
  int wave = threadIdx.x >> 5, lane = threadIdx.x & 31;
  int m0 = blockIdx.y * 64;
  int n0 = blockIdx.x * 256 + wave * 64;
  v8f acc[4][4];
#pragma unroll
  for (int i = 0; i < 4; ++i)
#pragma unroll
    for (int j = 0; j < 4; ++j) acc[i][j] = zero_v8f();
  for (int kt = 0; kt < H2 / 32; ++kt) {
    int k0 = kt * 32;
    v16bf a[4];
#pragma unroll
    for (int i = 0; i < 4; ++i) a[i] = load_frag(A, m0 + i * 16, k0, H2);
#pragma unroll
    for (int j = 0; j < 4; ++j) {
      v16bf bm = load_frag(Bm, n0 + j * 16, k0, H2);
#pragma unroll
      for (int i = 0; i < 4; ++i)
        acc[i][j] = __builtin_amdgcn_wmma_f32_16x16x32_bf16(false, a[i], false, bm, (short)0,
                                                            acc[i][j], false, false);
    }
  }
  int sel = lane >> 4, col = lane & 15;
#pragma unroll
  for (int j = 0; j < 4; ++j) {
    int n = n0 + j * 16 + col;
#pragma unroll
    for (int i = 0; i < 4; ++i) {
#pragma unroll
      for (int e = 0; e < 8; ++e) {
        int m = m0 + i * 16 + e + sel * 8;         // m = batch index
        pred[((size_t)m * Rn + r) * H2 + n] = acc[i][j][e];
      }
    }
  }
}

// ---------- dynamic routing (3 iters), one workgroup per batch element ----------
__global__ __launch_bounds__(256)
void routing_kernel(const float* __restrict__ pred, float* __restrict__ routes_out,
                    float* __restrict__ cls_out) {
  __shared__ float sp[Rn][Cn][Pn];   // 32 KB
  __shared__ float lg[Rn][Cn];
  __shared__ float rt[Rn][Cn];
  __shared__ float vv[Cn][Pn];
  __shared__ float scale_s[Cn];
  int b = blockIdx.x, tid = threadIdx.x;
  const float* pb = pred + (size_t)b * Rn * Cn * Pn;
  for (int i = tid; i < Rn * Cn * Pn; i += 256) ((float*)sp)[i] = pb[i];
  for (int i = tid; i < Rn * Cn; i += 256) ((float*)lg)[i] = 0.f;
  __syncthreads();
  int wv = tid >> 5, lane = tid & 31;              // wave wv owns row r = wv
  for (int it = 0; it < 3; ++it) {
    float x0 = lg[wv][lane], x1 = lg[wv][lane + 32];
    float m = fmaxf(x0, x1);
    for (int s = 16; s; s >>= 1) m = fmaxf(m, __shfl_xor(m, s, 32));
    float e0 = __expf(x0 - m), e1 = __expf(x1 - m);
    float sm = e0 + e1;
    for (int s = 16; s; s >>= 1) sm += __shfl_xor(sm, s, 32);
    rt[wv][lane] = e0 / sm; rt[wv][lane + 32] = e1 / sm;
    __syncthreads();
    for (int e = tid; e < Cn * Pn; e += 256) {
      int c = e >> 4, p = e & 15;
      float s = 0.f;
#pragma unroll
      for (int r = 0; r < Rn; ++r) s += rt[r][c] * sp[r][c][p];
      vv[c][p] = s;
    }
    __syncthreads();
    if (tid < Cn) {
      float sq = 0.f;
#pragma unroll
      for (int p = 0; p < Pn; ++p) sq += vv[tid][p] * vv[tid][p];
      scale_s[tid] = sq / ((1.f + sq) * sqrtf(sq + 1e-9f));
    }
    __syncthreads();
    for (int e = tid; e < Cn * Pn; e += 256) { int c = e >> 4; vv[c][e & 15] *= scale_s[c]; }
    __syncthreads();
    if (it < 2) {
      for (int e = tid; e < Rn * Cn; e += 256) {
        int r = e >> 6, c = e & 63;
        float d = 0.f;
#pragma unroll
        for (int p = 0; p < Pn; ++p) d += sp[r][c][p] * vv[c][p];
        lg[r][c] += d;
      }
      __syncthreads();
    }
  }
  for (int e = tid; e < Rn * Cn; e += 256)
    routes_out[(size_t)b * Rn * Cn + e] = rt[e >> 6][e & 63];
  if (tid < Cn) {
    float sq = 0.f;
#pragma unroll
    for (int p = 0; p < Pn; ++p) sq += vv[tid][p] * vv[tid][p];
    cls_out[(size_t)b * Cn + tid] = sqrtf(sq);
  }
}

// ---------- host ----------
extern "C" void kernel_launch(void* const* d_in, const int* in_sizes, int n_in,
                              void* d_out, int out_size, void* d_ws, size_t ws_size,
                              hipStream_t stream) {
  (void)in_sizes; (void)n_in; (void)out_size; (void)ws_size;
  const int*   ix    = (const int*)d_in[0];
  const int*   lens  = (const int*)d_in[1];
  const float* emb   = (const float*)d_in[2];
  const float* wih_f = (const float*)d_in[3];
  const float* whh_f = (const float*)d_in[4];
  const float* bih_f = (const float*)d_in[5];
  const float* bhh_f = (const float*)d_in[6];
  const float* wih_b = (const float*)d_in[7];
  const float* whh_b = (const float*)d_in[8];
  const float* bih_b = (const float*)d_in[9];
  const float* bhh_b = (const float*)d_in[10];
  const float* ws1   = (const float*)d_in[11];
  const float* ws2   = (const float*)d_in[12];
  const float* caps  = (const float*)d_in[13];

  float* out     = (float*)d_out;
  float* att_o   = out;                         // (256,8,64)
  float* cls_o   = out + 131072;                // (256,64)
  float* pred_o  = out + 147456;                // (256,8,64,16)
  float* route_o = out + 2244608;               // (256,8,64)

  char* wsb = (char*)d_ws;
  size_t off = 0;
  auto take = [&](size_t bytes) -> char* {
    char* p = wsb + off;
    off += (bytes + 255) & ~(size_t)255;
    return p;
  };
  bf16*  Xf    = (bf16*)take((size_t)Tn * Bn * Dp * 2);
  bf16*  Xb    = (bf16*)take((size_t)Tn * Bn * Dp * 2);
  bf16*  Wf    = (bf16*)take((size_t)G4H * KCAT * 2);
  bf16*  Wb    = (bf16*)take((size_t)G4H * KCAT * 2);
  float* biasF = (float*)take((size_t)G4H * 4);
  float* biasB = (float*)take((size_t)G4H * 4);
  bf16*  W1b   = (bf16*)take((size_t)DAn * H2 * 2);
  bf16*  capsT = (bf16*)take((size_t)Rn * H2 * H2 * 2);
  bf16*  hfb   = (bf16*)take((size_t)Bn * Hn * 2);
  bf16*  hbb   = (bf16*)take((size_t)Bn * Hn * 2);
  float* cf    = (float*)take((size_t)Bn * Hn * 4);
  float* cb    = (float*)take((size_t)Bn * Hn * 4);
  float* Zs    = (float*)take((size_t)Bn * G4H * 4);
  bf16*  outb  = (bf16*)take((size_t)Bn * Tn * H2 * 2);
  bf16*  U     = (bf16*)take((size_t)Bn * Tn * DAn * 2);
  float* sco   = (float*)take((size_t)Bn * Tn * Rn * 4);
  bf16*  semb  = (bf16*)take((size_t)Rn * Bn * H2 * 2);

  // prep
  embed_gather<<<(Tn * Bn * Dp + 255) / 256, 256, 0, stream>>>(ix, lens, emb, Xf, 0);
  embed_gather<<<(Tn * Bn * Dp + 255) / 256, 256, 0, stream>>>(ix, lens, emb, Xb, 1);
  pack_wcat<<<(G4H * KCAT + 255) / 256, 256, 0, stream>>>(wih_f, whh_f, bih_f, bhh_f, Wf, biasF);
  pack_wcat<<<(G4H * KCAT + 255) / 256, 256, 0, stream>>>(wih_b, whh_b, bih_b, bhh_b, Wb, biasB);
  pack_ws1<<<(DAn * H2 + 255) / 256, 256, 0, stream>>>(ws1, W1b);
  caps_transpose<<<(Rn * H2 * H2 + 255) / 256, 256, 0, stream>>>(caps, capsT);
  zero_state<<<(Bn * Hn + 255) / 256, 256, 0, stream>>>(hfb, hbb, cf, cb);

  // recurrent LSTM, forward then backward (64 WMMA GEMM steps each)
  dim3 gGrid(G4H / 256, Bn / 32);
  for (int t = 0; t < Tn; ++t) {
    lstm_step_gemm<<<gGrid, 128, 0, stream>>>(Xf + (size_t)t * Bn * Dp, hfb, Wf, biasF, Zs);
    lstm_gate<<<(Bn * Hn) / 256, 256, 0, stream>>>(Zs, lens, cf, hfb, outb, t, 0);
  }
  for (int t = 0; t < Tn; ++t) {
    lstm_step_gemm<<<gGrid, 128, 0, stream>>>(Xb + (size_t)t * Bn * Dp, hbb, Wb, biasB, Zs);
    lstm_gate<<<(Bn * Hn) / 256, 256, 0, stream>>>(Zs, lens, cb, hbb, outb, t, 1);
  }

  // attention
  attn_proj_gemm<<<(Bn * Tn) / 64, 128, 0, stream>>>(outb, W1b, U);
  score_kernel<<<(Bn * Tn * Rn) / 256, 256, 0, stream>>>(U, ws2, sco);
  att_softmax<<<(Bn * Rn) / 8, 256, 0, stream>>>(sco, att_o);
  sem_kernel<<<Bn * Rn, 256, 0, stream>>>(att_o, outb, semb);

  // capsule projection + routing
  caps_gemm<<<dim3(H2 / 256, Bn / 64, Rn), 128, 0, stream>>>(semb, capsT, pred_o);
  routing_kernel<<<Bn, 256, 0, stream>>>(pred_o, route_o, cls_o);
}
